// MoeLayer_50216757624886
// MI455X (gfx1250) — matrix operand order repaired and checked by reading, not compiled, not run on hardware
//
#include <hip/hip_runtime.h>

// MoE top-2 routed SwiGLU FFN for gfx1250 (MI455X).
// - wave32 + v_wmma_f32_16x16x32_f16, fp32 accuracy via f16 hi/lo split (3-term products)
// - grouped per-expert GEMMs over routed token lists (1/4 of the dense FLOPs)
// - double-buffered LDS k-loop; A tiles staged with GLOBAL_LOAD_ASYNC_TO_LDS_B128 (ASYNCcnt)

#define TOKS  16384
#define DIM   1024
#define HID   2048
#define NE    8
#define CAP   16384     // worst-case tokens per expert
#define LDA   40        // padded LDS row stride (halves) for A tiles
#define LDB   40        // padded LDS row stride (halves) for B tiles

typedef _Float16 v16h __attribute__((ext_vector_type(16)));
typedef _Float16 v8h  __attribute__((ext_vector_type(8)));
typedef _Float16 v4h  __attribute__((ext_vector_type(4)));
typedef float    v8f  __attribute__((ext_vector_type(8)));

#define WMMA_F16(a, b, c) \
  __builtin_amdgcn_wmma_f32_16x16x32_f16(false, (a), false, (b), (short)0, (c), false, false)

// ---- CDNA5 async global->LDS (ASYNCcnt-tracked, bypasses VGPRs) ----
__device__ __forceinline__ void async_ld_b128(unsigned lds_byte_off, const void* gaddr) {
  asm volatile("global_load_async_to_lds_b128 %0, %1, off"
               :: "v"(lds_byte_off), "v"(gaddr) : "memory");
}
__device__ __forceinline__ void wait_asynccnt0() {
#if __has_builtin(__builtin_amdgcn_s_wait_asynccnt)
  __builtin_amdgcn_s_wait_asynccnt(0);
#else
  asm volatile("s_wait_asynccnt 0" ::: "memory");
#endif
}
__device__ __forceinline__ unsigned lds_off(const void* p) {
  return (unsigned)(uintptr_t)p;   // low 32 bits of flat LDS address = wave LDS offset
}

// ---- workspace layout (all offsets 256B aligned) ----
static constexpr size_t OFF_CNT = 0;                                   // NE ints
static constexpr size_t OFF_TOK = 256;                                 // NE*CAP int
static constexpr size_t OFF_WL  = OFF_TOK + (size_t)NE * CAP * 4;      // NE*CAP float
static constexpr size_t OFF_XH  = OFF_WL  + (size_t)NE * CAP * 4;      // T*DIM f16
static constexpr size_t OFF_XL  = OFF_XH  + (size_t)TOKS * DIM * 2;
static constexpr size_t OFF_HH  = OFF_XL  + (size_t)TOKS * DIM * 2;    // CAP*HID f16
static constexpr size_t OFF_HL  = OFF_HH  + (size_t)CAP * HID * 2;
static constexpr size_t WS_NEED = OFF_HL  + (size_t)CAP * HID * 2;

// ---------------- fragment loaders (LDS -> VGPR, per ISA 7.12.2 layouts) -------------
// A: 16x32 f16, row-major in LDS [m][k], stride LDA halves.
__device__ __forceinline__ v16h frag_a(const _Float16* base, int lane) {
  const int m = lane & 15;
  const int g = lane >> 4;
  const v8h x0 = *(const v8h*)(base + m * LDA + g * 8);
  const v8h x1 = *(const v8h*)(base + m * LDA + 16 + g * 8);
  v16h r;
#pragma unroll
  for (int i = 0; i < 8; ++i) { r[i] = x0[i]; r[i + 8] = x1[i]; }
  return r;
}

// B: 32x16 f16 (KxN), stored in LDS as [n][k], stride LDB halves.
__device__ __forceinline__ v16h frag_b(const _Float16* base, int lane) {
  const int n = lane & 15;
  const int g = lane >> 4;
  const v8h x0 = *(const v8h*)(base + n * LDB + g * 8);
  const v8h x1 = *(const v8h*)(base + n * LDB + 16 + g * 8);
  v16h r;
#pragma unroll
  for (int i = 0; i < 8; ++i) { r[i] = x0[i]; r[i + 8] = x1[i]; }
  return r;
}

// ---------------- x -> (hi, lo) f16 split ----------------
__global__ __launch_bounds__(256) void moe_split_x(const float* __restrict__ x,
                                                   _Float16* __restrict__ xh,
                                                   _Float16* __restrict__ xl) {
  const size_t i = (size_t)blockIdx.x * 256 + threadIdx.x;   // handles 4 floats
  const float4 v = ((const float4*)x)[i];
  const float f[4] = {v.x, v.y, v.z, v.w};
  v4h hi, lo;
#pragma unroll
  for (int j = 0; j < 4; ++j) {
    const _Float16 h = (_Float16)f[j];
    hi[j] = h;
    lo[j] = (_Float16)(f[j] - (float)h);
  }
  *(v4h*)(xh + i * 4) = hi;
  *(v4h*)(xl + i * 4) = lo;
}

// ---------------- gating: logits, top-2, softmax, expert list build ----------------
__global__ __launch_bounds__(256) void moe_gate(const float* __restrict__ x,
                                                const float* __restrict__ wg,
                                                int* __restrict__ cnt,
                                                int* __restrict__ tok,
                                                float* __restrict__ wl) {
  const int lane = threadIdx.x & 31;
  const int t = blockIdx.x * 8 + (threadIdx.x >> 5);     // one wave32 per token
  const float* xr = x + (size_t)t * DIM;
  float acc[NE];
#pragma unroll
  for (int e = 0; e < NE; ++e) acc[e] = 0.f;
  for (int d = lane; d < DIM; d += 32) {
    const float xv = xr[d];
    const float4 g0 = *(const float4*)(wg + (size_t)d * NE);
    const float4 g1 = *(const float4*)(wg + (size_t)d * NE + 4);
    acc[0] += xv * g0.x; acc[1] += xv * g0.y; acc[2] += xv * g0.z; acc[3] += xv * g0.w;
    acc[4] += xv * g1.x; acc[5] += xv * g1.y; acc[6] += xv * g1.z; acc[7] += xv * g1.w;
  }
#pragma unroll
  for (int off = 16; off > 0; off >>= 1)
#pragma unroll
    for (int e = 0; e < NE; ++e) acc[e] += __shfl_xor(acc[e], off, 32);

  if (lane == 0) {
    int e0 = 0; float v0 = acc[0];
#pragma unroll
    for (int e = 1; e < NE; ++e) if (acc[e] > v0) { v0 = acc[e]; e0 = e; }
    int e1 = (e0 == 0) ? 1 : 0; float v1 = acc[e1];
#pragma unroll
    for (int e = 0; e < NE; ++e) if (e != e0 && acc[e] > v1) { v1 = acc[e]; e1 = e; }
    const float p = __expf(v1 - v0);          // v0 is the max
    const float s = 1.f + p;
    const int i0 = atomicAdd(&cnt[e0], 1);
    tok[e0 * CAP + i0] = t; wl[e0 * CAP + i0] = 1.f / s;
    const int i1 = atomicAdd(&cnt[e1], 1);
    tok[e1 * CAP + i1] = t; wl[e1 * CAP + i1] = p / s;
  }
}

// ---------------- FFN1: h = silu(x w1^T) * (x w3^T), rows gathered per expert ------
__global__ __launch_bounds__(256) void moe_ffn1(const _Float16* __restrict__ xh,
                                                const _Float16* __restrict__ xl,
                                                const float* __restrict__ w1e,
                                                const float* __restrict__ w3e,
                                                const int* __restrict__ cnt,
                                                const int* __restrict__ tok, int e,
                                                _Float16* __restrict__ hh,
                                                _Float16* __restrict__ hl) {
  const int rows = cnt[e];
  const int m0 = blockIdx.y * 128;
  if (m0 >= rows) return;
  const int n0 = blockIdx.x * 64;

  __shared__ int s_tok[128];
  __shared__ __align__(16) _Float16 sAh[2][128][LDA], sAl[2][128][LDA];
  __shared__ __align__(16) _Float16 sB1h[2][64][LDB], sB1l[2][64][LDB];
  __shared__ __align__(16) _Float16 sB3h[2][64][LDB], sB3l[2][64][LDB];

  const int tid = threadIdx.x;
  const int lane = tid & 31;
  const int wv = tid >> 5;
  const int wm = wv >> 1;      // 0..3 -> 32-row strip
  const int wn = wv & 1;       // 0..1 -> 32-col strip

  if (tid < 128) {
    int idx = m0 + tid; if (idx >= rows) idx = rows - 1;
    s_tok[tid] = tok[(size_t)e * CAP + idx];
  }
  __syncthreads();

  float4 rb1[2], rb3[2];   // B prefetch registers

  auto stageA = [&](int buf, int k0) {   // async global->LDS, no VGPR round-trip
#pragma unroll
    for (int t2 = 0; t2 < 2; ++t2) {
      const int i = tid + t2 * 256;      // 512 x 16B segments per array
      const int r = i >> 2, j = i & 3;
      const size_t src = (size_t)s_tok[r] * DIM + (k0 + j * 8);
      async_ld_b128(lds_off(&sAh[buf][r][j * 8]), xh + src);
      async_ld_b128(lds_off(&sAl[buf][r][j * 8]), xl + src);
    }
  };
  auto loadB = [&](int k0) {
#pragma unroll
    for (int t2 = 0; t2 < 2; ++t2) {
      const int i = tid + t2 * 256;
      const int n = i >> 3, j = i & 7;
      const size_t src = (size_t)(n0 + n) * DIM + (k0 + j * 4);
      rb1[t2] = *(const float4*)(w1e + src);
      rb3[t2] = *(const float4*)(w3e + src);
    }
  };
  auto storeB = [&](int buf) {           // fp32 -> f16 hi/lo split into LDS
#pragma unroll
    for (int t2 = 0; t2 < 2; ++t2) {
      const int i = tid + t2 * 256;
      const int n = i >> 3, j = i & 7;
      const float fa[4] = {rb1[t2].x, rb1[t2].y, rb1[t2].z, rb1[t2].w};
      const float fb[4] = {rb3[t2].x, rb3[t2].y, rb3[t2].z, rb3[t2].w};
#pragma unroll
      for (int q = 0; q < 4; ++q) {
        const _Float16 h1 = (_Float16)fa[q];
        sB1h[buf][n][j * 4 + q] = h1;
        sB1l[buf][n][j * 4 + q] = (_Float16)(fa[q] - (float)h1);
        const _Float16 h3 = (_Float16)fb[q];
        sB3h[buf][n][j * 4 + q] = h3;
        sB3l[buf][n][j * 4 + q] = (_Float16)(fb[q] - (float)h3);
      }
    }
  };

  const v8f zero = {0.f, 0.f, 0.f, 0.f, 0.f, 0.f, 0.f, 0.f};
  v8f acc1[2][2], acc3[2][2];
#pragma unroll
  for (int s = 0; s < 2; ++s)
#pragma unroll
    for (int u = 0; u < 2; ++u) { acc1[s][u] = zero; acc3[s][u] = zero; }

  // prologue: stage chunk 0
  stageA(0, 0);
  loadB(0);
  storeB(0);
  wait_asynccnt0();
  __syncthreads();

  int cur = 0;
  for (int k0 = 0; k0 < DIM; k0 += 32) {
    const int nxt = k0 + 32;
    if (nxt < DIM) { stageA(cur ^ 1, nxt); loadB(nxt); }   // overlap with compute

    const v16h ah0 = frag_a(&sAh[cur][wm * 32][0], lane);
    const v16h ah1 = frag_a(&sAh[cur][wm * 32 + 16][0], lane);
    const v16h al0 = frag_a(&sAl[cur][wm * 32][0], lane);
    const v16h al1 = frag_a(&sAl[cur][wm * 32 + 16][0], lane);
#pragma unroll
    for (int u = 0; u < 2; ++u) {
      const int nb = wn * 32 + u * 16;
      v16h bh = frag_b(&sB1h[cur][nb][0], lane);
      v16h bl = frag_b(&sB1l[cur][nb][0], lane);
      acc1[0][u] = WMMA_F16(ah0, bh, acc1[0][u]);
      acc1[0][u] = WMMA_F16(ah0, bl, acc1[0][u]);
      acc1[0][u] = WMMA_F16(al0, bh, acc1[0][u]);
      acc1[1][u] = WMMA_F16(ah1, bh, acc1[1][u]);
      acc1[1][u] = WMMA_F16(ah1, bl, acc1[1][u]);
      acc1[1][u] = WMMA_F16(al1, bh, acc1[1][u]);
      bh = frag_b(&sB3h[cur][nb][0], lane);
      bl = frag_b(&sB3l[cur][nb][0], lane);
      acc3[0][u] = WMMA_F16(ah0, bh, acc3[0][u]);
      acc3[0][u] = WMMA_F16(ah0, bl, acc3[0][u]);
      acc3[0][u] = WMMA_F16(al0, bh, acc3[0][u]);
      acc3[1][u] = WMMA_F16(ah1, bh, acc3[1][u]);
      acc3[1][u] = WMMA_F16(ah1, bl, acc3[1][u]);
      acc3[1][u] = WMMA_F16(al1, bh, acc3[1][u]);
    }

    if (nxt < DIM) storeB(cur ^ 1);
    wait_asynccnt0();
    __syncthreads();
    cur ^= 1;
  }

  // epilogue: fp32 SwiGLU, store h as split f16. C layout: VGPR r -> M=r(+8), N=lane&15.
  const int mbase = m0 + wm * 32;
  const int cbase = n0 + wn * 32 + (lane & 15);
  const int moff = (lane >> 4) * 8;
#pragma unroll
  for (int s = 0; s < 2; ++s)
#pragma unroll
    for (int u = 0; u < 2; ++u)
#pragma unroll
      for (int r = 0; r < 8; ++r) {
        const int gr = mbase + s * 16 + moff + r;
        if (gr < rows) {
          const int col = cbase + u * 16;
          const float a1v = acc1[s][u][r];
          const float hval = (a1v / (1.f + __expf(-a1v))) * acc3[s][u][r];
          const _Float16 hi = (_Float16)hval;
          hh[(size_t)gr * HID + col] = hi;
          hl[(size_t)gr * HID + col] = (_Float16)(hval - (float)hi);
        }
      }
}

// ---------------- FFN2: y += weight * (h @ w2[e]), scatter-add over tokens --------
__global__ __launch_bounds__(256) void moe_ffn2(const _Float16* __restrict__ hh,
                                                const _Float16* __restrict__ hl,
                                                const float* __restrict__ w2e,
                                                const int* __restrict__ cnt,
                                                const int* __restrict__ tok,
                                                const float* __restrict__ wl, int e,
                                                float* __restrict__ y) {
  const int rows = cnt[e];
  const int m0 = blockIdx.y * 128;
  if (m0 >= rows) return;
  const int n0 = blockIdx.x * 64;

  __shared__ int s_tk[128];
  __shared__ float s_wt[128];
  __shared__ __align__(16) _Float16 sAh[2][128][LDA], sAl[2][128][LDA];
  __shared__ __align__(16) _Float16 sBh[2][64][LDB], sBl[2][64][LDB];

  const int tid = threadIdx.x;
  const int lane = tid & 31;
  const int wv = tid >> 5;
  const int wm = wv >> 1;
  const int wn = wv & 1;

  if (tid < 128) {
    int idx = m0 + tid; if (idx >= rows) idx = rows - 1;
    s_tk[tid] = tok[(size_t)e * CAP + idx];
    s_wt[tid] = wl[(size_t)e * CAP + idx];
  }
  __syncthreads();

  float rb2[8];   // B prefetch registers (w2 tile, 2048 floats / 256 threads)

  auto stageA = [&](int buf, int k0) {
#pragma unroll
    for (int t2 = 0; t2 < 2; ++t2) {
      const int i = tid + t2 * 256;
      const int r = i >> 2, j = i & 3;
      int gr = m0 + r; if (gr >= rows) gr = rows - 1;
      const size_t src = (size_t)gr * HID + (k0 + j * 8);
      async_ld_b128(lds_off(&sAh[buf][r][j * 8]), hh + src);
      async_ld_b128(lds_off(&sAl[buf][r][j * 8]), hl + src);
    }
  };
  auto loadB = [&](int k0) {
#pragma unroll
    for (int t2 = 0; t2 < 8; ++t2) {
      const int i = tid + t2 * 256;
      const int k = i >> 6, n = i & 63;
      rb2[t2] = w2e[(size_t)(k0 + k) * DIM + n0 + n];
    }
  };
  auto storeB = [&](int buf) {           // transpose into [n][k] + hi/lo split
#pragma unroll
    for (int t2 = 0; t2 < 8; ++t2) {
      const int i = tid + t2 * 256;
      const int k = i >> 6, n = i & 63;
      const _Float16 hi = (_Float16)rb2[t2];
      sBh[buf][n][k] = hi;
      sBl[buf][n][k] = (_Float16)(rb2[t2] - (float)hi);
    }
  };

  const v8f zero = {0.f, 0.f, 0.f, 0.f, 0.f, 0.f, 0.f, 0.f};
  v8f acc[2][2];
#pragma unroll
  for (int s = 0; s < 2; ++s)
#pragma unroll
    for (int u = 0; u < 2; ++u) acc[s][u] = zero;

  stageA(0, 0);
  loadB(0);
  storeB(0);
  wait_asynccnt0();
  __syncthreads();

  int cur = 0;
  for (int k0 = 0; k0 < HID; k0 += 32) {
    const int nxt = k0 + 32;
    if (nxt < HID) { stageA(cur ^ 1, nxt); loadB(nxt); }

    const v16h ah0 = frag_a(&sAh[cur][wm * 32][0], lane);
    const v16h ah1 = frag_a(&sAh[cur][wm * 32 + 16][0], lane);
    const v16h al0 = frag_a(&sAl[cur][wm * 32][0], lane);
    const v16h al1 = frag_a(&sAl[cur][wm * 32 + 16][0], lane);
#pragma unroll
    for (int u = 0; u < 2; ++u) {
      const int nb = wn * 32 + u * 16;
      const v16h bh = frag_b(&sBh[cur][nb][0], lane);
      const v16h bl = frag_b(&sBl[cur][nb][0], lane);
      acc[0][u] = WMMA_F16(ah0, bh, acc[0][u]);
      acc[0][u] = WMMA_F16(ah0, bl, acc[0][u]);
      acc[0][u] = WMMA_F16(al0, bh, acc[0][u]);
      acc[1][u] = WMMA_F16(ah1, bh, acc[1][u]);
      acc[1][u] = WMMA_F16(ah1, bl, acc[1][u]);
      acc[1][u] = WMMA_F16(al1, bh, acc[1][u]);
    }

    if (nxt < HID) storeB(cur ^ 1);
    wait_asynccnt0();
    __syncthreads();
    cur ^= 1;
  }

  // epilogue: weighted scatter-add into y (exactly 2 commutative adds per element)
  const int cbase = n0 + wn * 32 + (lane & 15);
  const int moff = (lane >> 4) * 8;
#pragma unroll
  for (int s = 0; s < 2; ++s)
#pragma unroll
    for (int u = 0; u < 2; ++u)
#pragma unroll
      for (int r = 0; r < 8; ++r) {
        const int lr = wm * 32 + s * 16 + moff + r;
        const int gr = m0 + lr;
        if (gr < rows) {
          atomicAdd(&y[(size_t)s_tk[lr] * DIM + cbase + u * 16], s_wt[lr] * acc[s][u][r]);
        }
      }
}

// ---------------- host entry ----------------
extern "C" void kernel_launch(void* const* d_in, const int* in_sizes, int n_in,
                              void* d_out, int out_size, void* d_ws, size_t ws_size,
                              hipStream_t stream) {
  (void)in_sizes; (void)n_in; (void)out_size;
  if (ws_size < WS_NEED) return;

  const float* x  = (const float*)d_in[0];
  const float* wg = (const float*)d_in[1];
  const float* w1 = (const float*)d_in[2];
  const float* w2 = (const float*)d_in[3];
  const float* w3 = (const float*)d_in[4];
  float* y = (float*)d_out;

  char* ws = (char*)d_ws;
  int*      cnt = (int*)(ws + OFF_CNT);
  int*      tok = (int*)(ws + OFF_TOK);
  float*    wl  = (float*)(ws + OFF_WL);
  _Float16* xh  = (_Float16*)(ws + OFF_XH);
  _Float16* xl  = (_Float16*)(ws + OFF_XL);
  _Float16* hh  = (_Float16*)(ws + OFF_HH);
  _Float16* hl  = (_Float16*)(ws + OFF_HL);

  hipMemsetAsync(d_out, 0, (size_t)TOKS * DIM * sizeof(float), stream);
  hipMemsetAsync(cnt, 0, NE * sizeof(int), stream);

  moe_split_x<<<(TOKS * DIM / 4) / 256, 256, 0, stream>>>(x, xh, xl);
  moe_gate<<<TOKS / 8, 256, 0, stream>>>(x, wg, cnt, tok, wl);

  for (int e = 0; e < NE; ++e) {
    const float* w1e = w1 + (size_t)e * HID * DIM;
    const float* w2e = w2 + (size_t)e * HID * DIM;
    const float* w3e = w3 + (size_t)e * HID * DIM;
    moe_ffn1<<<dim3(HID / 64, CAP / 128), 256, 0, stream>>>(xh, xl, w1e, w3e, cnt, tok, e, hh, hl);
    moe_ffn2<<<dim3(DIM / 64, CAP / 128), 256, 0, stream>>>(hh, hl, w2e, cnt, tok, wl, e, y);
  }
}
